// ConvNeXtSynthesisLayer_48060684042363
// MI455X (gfx1250) — matrix operand order repaired and checked by reading, not compiled, not run on hardware
//
#include <hip/hip_runtime.h>
#include <math.h>

typedef _Float16 h16;
typedef __attribute__((ext_vector_type(16))) _Float16 v16h;
typedef __attribute__((ext_vector_type(8)))  float    v8f;

union AFrag { v16h h; uint4 u[2]; };
union Pack8 { h16 h[8]; uint4 u; };
union Pack16 { h16 h[16]; uint4 u[2]; };

#define B_   16
#define C_   256
#define HW_  4096
#define C4_  1024
#define WD_  512
#define G_   32

// ---------------------------------------------------------------------------
// small helpers
// ---------------------------------------------------------------------------
__global__ __launch_bounds__(256) void zero_kernel(float* p, int n) {
    int i = blockIdx.x * 256 + threadIdx.x;
    if (i < n) p[i] = 0.0f;
}

__global__ __launch_bounds__(256) void cvt_f16_kernel(const float* __restrict__ src,
                                                      h16* __restrict__ dst, int n) {
    int i = blockIdx.x * 256 + threadIdx.x;
    if (i < n) dst[i] = (h16)src[i];
}

// ---------------------------------------------------------------------------
// style: s = w @ aff_w.T + aff_b ; style = s1*s2 + s3   (B x C)
// ---------------------------------------------------------------------------
__global__ __launch_bounds__(256) void style_kernel(const float* __restrict__ w,
                                                    const float* __restrict__ aff_w,
                                                    const float* __restrict__ aff_b,
                                                    float* __restrict__ style) {
    int idx = blockIdx.x * 256 + threadIdx.x;      // 0..4095
    int b = idx >> 8, c = idx & 255;
    const float4* wv = (const float4*)(w + (size_t)b * WD_);
    float s[3];
#pragma unroll
    for (int j = 0; j < 3; ++j) {
        const float4* av = (const float4*)(aff_w + (size_t)(j * C_ + c) * WD_);
        float acc = 0.0f;
        for (int k = 0; k < WD_ / 4; ++k) {
            float4 a = av[k], ww = wv[k];
            acc += a.x * ww.x + a.y * ww.y + a.z * ww.z + a.w * ww.w;
        }
        s[j] = acc + aff_b[j * C_ + c];
    }
    style[b * C_ + c] = s[0] * s[1] + s[2];
}

// ---------------------------------------------------------------------------
// depthwise 7x7 conv (same padding) -> f16 y [b][c][hw]; f32 group stats
// block = (b, c, 4-row strip of H); 256 threads, 1 output pixel each
// ---------------------------------------------------------------------------
__global__ __launch_bounds__(256) void dwconv_kernel(const float* __restrict__ x,
                                                     const float* __restrict__ dw_w,
                                                     const float* __restrict__ dw_b,
                                                     h16* __restrict__ y,
                                                     float* __restrict__ stats) {
    int bid = blockIdx.x;                 // B*C*16 = 65536
    int hs = bid & 15;
    int c  = (bid >> 4) & 255;
    int b  = bid >> 12;
    int h0 = hs * 4;

    __shared__ float  tile[700];          // 10 x 70 input halo tile
    __shared__ float  wk[49];
    __shared__ float2 red[256];

    int t = threadIdx.x;
    const float* xc = x + ((size_t)(b * C_ + c)) * HW_;
    for (int i = t; i < 700; i += 256) {
        int r  = i / 70;
        int cc = i % 70 - 3;
        int hh = h0 - 3 + r;
        float v = 0.0f;
        if (hh >= 0 && hh < 64 && cc >= 0 && cc < 64) v = xc[hh * 64 + cc];
        tile[i] = v;
    }
    if (t < 49) wk[t] = dw_w[c * 49 + t];
    __syncthreads();

    int ww = t & 63, hh = t >> 6;
    float acc = dw_b[c];
#pragma unroll
    for (int ky = 0; ky < 7; ++ky)
#pragma unroll
        for (int kx = 0; kx < 7; ++kx)
            acc += wk[ky * 7 + kx] * tile[(hh + ky) * 70 + ww + kx];

    y[((size_t)(b * C_ + c)) * HW_ + h0 * 64 + t] = (h16)acc;

    red[t] = make_float2(acc, acc * acc);
    __syncthreads();
    for (int s = 128; s > 0; s >>= 1) {
        if (t < s) { red[t].x += red[t + s].x; red[t].y += red[t + s].y; }
        __syncthreads();
    }
    if (t == 0) {
        int g = c >> 3;                   // C/G = 8 channels per group
        atomicAdd(&stats[(b * G_ + g) * 2 + 0], red[0].x);
        atomicAdd(&stats[(b * G_ + g) * 2 + 1], red[0].y);
    }
}

__global__ __launch_bounds__(256) void finalize_stats(const float* __restrict__ stats,
                                                      float* __restrict__ mu,
                                                      float* __restrict__ rsig) {
    int idx = blockIdx.x * 256 + threadIdx.x;
    if (idx >= B_ * G_) return;
    const float n = 8.0f * 4096.0f;       // elems per (b,g)
    float s = stats[idx * 2], sq = stats[idx * 2 + 1];
    float m = s / n;
    float var = sq / n - m * m;
    mu[idx]   = m;
    rsig[idx] = rsqrtf(var + 1e-5f);
}

// ---------------------------------------------------------------------------
// prep: wmA[b,o,i] = pw1_w[o,i]*style[b,i]*dcoef[b,o] * (rsig*g)[b,i]  (f16)
//       bias2[b,o] = pw1_b[o] + sum_i wm_d[b,o,i] * (norm_b - mu*rsig*g)[b,i]
// block = (b,o), 256 threads (one per i)
// ---------------------------------------------------------------------------
__global__ __launch_bounds__(256) void prep_kernel(const float* __restrict__ pw1_w,
                                                   const float* __restrict__ pw1_b,
                                                   const float* __restrict__ style,
                                                   const float* __restrict__ norm_g,
                                                   const float* __restrict__ norm_b,
                                                   const float* __restrict__ mu,
                                                   const float* __restrict__ rsig,
                                                   h16* __restrict__ wmA,
                                                   float* __restrict__ bias2) {
    int o = blockIdx.x & (C4_ - 1);
    int b = blockIdx.x >> 10;
    int i = threadIdx.x;
    __shared__ float red[256];

    float t = pw1_w[o * C_ + i] * style[b * C_ + i];
    red[i] = t * t;
    __syncthreads();
    for (int s = 128; s > 0; s >>= 1) { if (i < s) red[i] += red[i + s]; __syncthreads(); }
    float dcoef = rsqrtf(red[0] + 1e-8f);
    __syncthreads();

    float wmd = t * dcoef;
    int g = i >> 3;
    float rs = rsig[b * G_ + g];
    float a  = rs * norm_g[i];
    float d  = norm_b[i] - mu[b * G_ + g] * a;
    wmA[((size_t)(b * C4_ + o)) * C_ + i] = (h16)(wmd * a);

    red[i] = wmd * d;
    __syncthreads();
    for (int s = 128; s > 0; s >>= 1) { if (i < s) red[i] += red[i + s]; __syncthreads(); }
    if (i == 0) bias2[b * C4_ + o] = pw1_b[o] + red[0];
}

// ---------------------------------------------------------------------------
// transpose f16 y[b][c][hw] -> yT[b][hw][c]  (64x64 LDS tiles)
// ---------------------------------------------------------------------------
__global__ __launch_bounds__(256) void transpose_kernel(const h16* __restrict__ y,
                                                        h16* __restrict__ yT) {
    int b = blockIdx.z, c0 = blockIdx.y * 64, hw0 = blockIdx.x * 64;
    __shared__ h16 tile[64 * 65];
    int t = threadIdx.x;
    const h16* yb = y + (size_t)b * (C_ * HW_);

    int cr = t >> 2, q16 = (t & 3) * 16;
    Pack16 ld;
    const h16* src = yb + (size_t)(c0 + cr) * HW_ + hw0 + q16;
    ld.u[0] = *(const uint4*)(src);
    ld.u[1] = *(const uint4*)(src + 8);
#pragma unroll
    for (int j = 0; j < 16; ++j) tile[cr * 65 + q16 + j] = ld.h[j];
    __syncthreads();

    int hr = t >> 2, c16 = (t & 3) * 16;
    Pack16 st;
#pragma unroll
    for (int j = 0; j < 16; ++j) st.h[j] = tile[(c16 + j) * 65 + hr];
    h16* dst = yT + (size_t)b * (C_ * HW_) + (size_t)(hw0 + hr) * C_ + c0 + c16;
    *(uint4*)(dst)     = st.u[0];
    *(uint4*)(dst + 8) = st.u[1];
}

// ---------------------------------------------------------------------------
// WMMA GEMM core (software-pipelined, double-buffered LDS B tile)
//   A:   [M][K] f16 row-major (K contiguous)
//   B:   [N][K] f16 (K contiguous per column n)
//   block tile 128x128; 8 waves * (16 rows x 8 wmma n-tiles); K-step 32
//   LDS row stride 80 B (16B-aligned, bank-conflict-free fragment reads)
// ---------------------------------------------------------------------------
#define GEMM_CORE(Abase, Bbase, Kdim)                                            \
    v8f acc[8];                                                                  \
    _Pragma("unroll")                                                            \
    for (int nt = 0; nt < 8; ++nt)                                               \
        _Pragma("unroll")                                                        \
        for (int r = 0; r < 8; ++r) acc[nt][r] = 0.0f;                           \
    const int ch0 = tid, ch1 = tid + 256;                                        \
    const int na = ch0 >> 2, qa = ch0 & 3;                                       \
    const int nb = ch1 >> 2, qb = ch1 & 3;                                       \
    const h16* Brow_a = (Bbase) + (size_t)(n0 + na) * (Kdim) + qa * 8;           \
    const h16* Brow_b = (Bbase) + (size_t)(n0 + nb) * (Kdim) + qb * 8;           \
    const h16* Arow   = (Abase) + (size_t)row * (Kdim);                          \
    /* prologue: stage k0 = 0 tile into buffer 0 */                              \
    sB[na * 5 + qa] = *(const uint4*)(Brow_a);                                   \
    sB[nb * 5 + qb] = *(const uint4*)(Brow_b);                                   \
    AFrag a_cur;                                                                 \
    a_cur.u[0] = *(const uint4*)(Arow + hb * 8);                                 \
    a_cur.u[1] = *(const uint4*)(Arow + hb * 8 + 16);                            \
    __syncthreads();                                                             \
    const int NK = (Kdim) / 32;                                                  \
    for (int ks = 0; ks < NK; ++ks) {                                            \
        const int cur   = ks & 1;                                                \
        const int knext = ((ks + 1) * 32) & ((Kdim) - 1); /* wraps, no branch */ \
        /* issue next-tile global loads + next A prefetch (latency hidden) */    \
        uint4 g0 = *(const uint4*)(Brow_a + knext);                              \
        uint4 g1 = *(const uint4*)(Brow_b + knext);                              \
        AFrag a_nxt;                                                             \
        a_nxt.u[0] = *(const uint4*)(Arow + knext + hb * 8);                     \
        a_nxt.u[1] = *(const uint4*)(Arow + knext + hb * 8 + 16);                \
        /* 8 WMMAs on current LDS buffer */                                      \
        const uint4* sBc = sB + cur * 640;                                       \
        _Pragma("unroll")                                                        \
        for (int nt = 0; nt < 8; ++nt) {                                         \
            AFrag bf;                                                            \
            int base = (nt * 16 + ln) * 5 + hb;                                  \
            bf.u[0] = sBc[base];                                                 \
            bf.u[1] = sBc[base + 2];                                             \
            acc[nt] = __builtin_amdgcn_wmma_f32_16x16x32_f16(                    \
                false, a_cur.h, false, bf.h, (short)0, acc[nt], false, false);   \
        }                                                                        \
        /* commit staged registers into the other buffer */                      \
        uint4* sBn = sB + (cur ^ 1) * 640;                                       \
        sBn[na * 5 + qa] = g0;                                                   \
        sBn[nb * 5 + qb] = g1;                                                   \
        __syncthreads();                                                         \
        a_cur = a_nxt;                                                           \
    }

// ---------------------------------------------------------------------------
// GEMM1: z[b][n][o] = GELU( wmA[b] (1024x256) @ yT[b]^T + bias2[b] ), f16 out
// ---------------------------------------------------------------------------
__global__ __launch_bounds__(256) void gemm1_mod_gelu(const h16* __restrict__ wmA,
                                                      const float* __restrict__ bias2,
                                                      const h16* __restrict__ yT,
                                                      h16* __restrict__ z) {
    const int b  = blockIdx.z;
    const int n0 = blockIdx.x * 128;
    const int m0 = blockIdx.y * 128;
    const int tid  = threadIdx.x;
    const int wave = tid >> 5;
    const int lane = tid & 31;
    const int hb   = lane >> 4;
    const int ln   = lane & 15;
    const int row  = m0 + wave * 16 + ln;

    const h16* Ab = wmA + (size_t)b * (C4_ * C_);
    const h16* Bb = yT  + (size_t)b * (HW_ * C_);   // [n][k], K = 256

    __shared__ uint4 sB[1280];      // 2 x (128 rows x 80 bytes)

    GEMM_CORE(Ab, Bb, C_)

    const int obase = m0 + wave * 16 + hb * 8;
    const float* b2 = bias2 + b * C4_ + obase;
    h16* zb = z + (size_t)b * ((size_t)HW_ * C4_);
#pragma unroll
    for (int nt = 0; nt < 8; ++nt) {
        int n = n0 + nt * 16 + ln;
        Pack8 p;
#pragma unroll
        for (int r = 0; r < 8; ++r) {
            float v = acc[nt][r] + b2[r];
            v = 0.5f * v * (1.0f + erff(v * 0.70710678118654752f));  // exact GELU
            p.h[r] = (h16)v;
        }
        *(uint4*)(zb + (size_t)n * C4_ + obase) = p.u;
    }
}

// ---------------------------------------------------------------------------
// GEMM2: out[b][o][n] = gamma[o]*(pw2 (256x1024) @ z[b] + pw2_b) + x[b][o][n]
// ---------------------------------------------------------------------------
__global__ __launch_bounds__(256) void gemm2_residual(const h16* __restrict__ pw2h,
                                                      const float* __restrict__ pw2_b,
                                                      const h16* __restrict__ z,
                                                      const float* __restrict__ gamma,
                                                      const float* __restrict__ x,
                                                      float* __restrict__ out) {
    const int b  = blockIdx.z;
    const int n0 = blockIdx.x * 128;
    const int m0 = blockIdx.y * 128;
    const int tid  = threadIdx.x;
    const int wave = tid >> 5;
    const int lane = tid & 31;
    const int hb   = lane >> 4;
    const int ln   = lane & 15;
    const int row  = m0 + wave * 16 + ln;

    const h16* Bb = z + (size_t)b * ((size_t)HW_ * C4_);  // [n][k], K = 1024

    __shared__ uint4 sB[1280];

    GEMM_CORE(pw2h, Bb, C4_)

    const int obase = m0 + wave * 16 + hb * 8;
    const float* pb = pw2_b + obase;
    const float* gm = gamma + obase;
    const float* xb = x   + ((size_t)b * C_ + obase) * HW_;
    float*       ob = out + ((size_t)b * C_ + obase) * HW_;
#pragma unroll
    for (int nt = 0; nt < 8; ++nt) {
        int n = n0 + nt * 16 + ln;
#pragma unroll
        for (int r = 0; r < 8; ++r) {
            float v = acc[nt][r] + pb[r];
            v = gm[r] * v + xb[(size_t)r * HW_ + n];
            ob[(size_t)r * HW_ + n] = v;
        }
    }
}

// ---------------------------------------------------------------------------
// host launcher
// ---------------------------------------------------------------------------
extern "C" void kernel_launch(void* const* d_in, const int* in_sizes, int n_in,
                              void* d_out, int out_size, void* d_ws, size_t ws_size,
                              hipStream_t stream) {
    (void)in_sizes; (void)n_in; (void)out_size; (void)ws_size;
    const float* x      = (const float*)d_in[0];
    const float* w      = (const float*)d_in[1];
    const float* aff_w  = (const float*)d_in[2];
    const float* aff_b  = (const float*)d_in[3];
    const float* dw_w   = (const float*)d_in[4];
    const float* dw_b   = (const float*)d_in[5];
    const float* norm_g = (const float*)d_in[6];
    const float* norm_b = (const float*)d_in[7];
    const float* pw1_w  = (const float*)d_in[8];
    const float* pw1_b  = (const float*)d_in[9];
    const float* pw2_w  = (const float*)d_in[10];
    const float* pw2_b  = (const float*)d_in[11];
    const float* gamma  = (const float*)d_in[12];
    float* out = (float*)d_out;

    // workspace layout (bytes); total required ~169 MB
    char* ws = (char*)d_ws;
    float* stats = (float*)(ws + 0);            //  1024 f32 (sum,sumsq per b,g)
    float* mu    = (float*)(ws + 4096);         //   512 f32
    float* rsig  = (float*)(ws + 6144);         //   512 f32
    float* style = (float*)(ws + 8192);         //  4096 f32
    float* bias2 = (float*)(ws + 24576);        // 16384 f32
    h16*   pw2h  = (h16*)  (ws + 90112);        //  256K f16
    h16*   wmA   = (h16*)  (ws + 614400);       //  4M f16 (8 MB)
    h16*   yT    = (h16*)  (ws + 9003008);      // 16M f16 (32 MB)  [b][hw][c]
    h16*   y     = (h16*)  (ws + 42557440);     // 16M f16 (32 MB)  [b][c][hw]
    h16*   z     = (h16*)  (ws + 42557440);     // 64M f16 (128 MB) reuses dead y

    zero_kernel   <<<4,     256, 0, stream>>>(stats, B_ * G_ * 2);
    style_kernel  <<<16,    256, 0, stream>>>(w, aff_w, aff_b, style);
    dwconv_kernel <<<65536, 256, 0, stream>>>(x, dw_w, dw_b, y, stats);
    finalize_stats<<<2,     256, 0, stream>>>(stats, mu, rsig);
    prep_kernel   <<<16384, 256, 0, stream>>>(pw1_w, pw1_b, style, norm_g, norm_b,
                                              mu, rsig, wmA, bias2);
    cvt_f16_kernel<<<1024,  256, 0, stream>>>(pw2_w, pw2h, C_ * C4_);
    transpose_kernel<<<dim3(64, 4, 16),  256, 0, stream>>>(y, yT);
    gemm1_mod_gelu  <<<dim3(32, 8, 16),  256, 0, stream>>>(wmA, bias2, yT, z);
    gemm2_residual  <<<dim3(32, 2, 16),  256, 0, stream>>>(pw2h, pw2_b, z, gamma, x, out);
}
